// HGRN2Attention_13752485282226
// MI455X (gfx1250) — matrix operand order repaired and checked by reading, not compiled, not run on hardware
//
#include <hip/hip_runtime.h>

// ---------------------------------------------------------------------------
// CDNA5 (gfx1250) HGRN2 forward: bf16 WMMA GEMMs + chunked WMMA linear-attn scan
// ---------------------------------------------------------------------------

typedef __attribute__((ext_vector_type(16))) __bf16 bf16x16;
typedef __attribute__((ext_vector_type(8)))  float  floatx8;
typedef __attribute__((ext_vector_type(4)))  int    i32x4;

#define WMMA_BF16(a, b, c) \
  __builtin_amdgcn_wmma_f32_16x16x32_bf16(false, (a), false, (b), (short)0, (c), false, false)

#if defined(__AMDGCN__) && __has_builtin(__builtin_amdgcn_global_load_async_to_lds_b128) && \
    __has_builtin(__builtin_amdgcn_s_wait_asynccnt)
#define HGRN2_ASYNC 1
#endif

static __device__ __forceinline__ int laneid() { return threadIdx.x & 31; }

// 16-byte global -> LDS copy: async-DMA (ASYNCcnt) on CDNA5, else plain copy.
static __device__ __forceinline__ void copy16(const void* g, void* l) {
#ifdef HGRN2_ASYNC
  __builtin_amdgcn_global_load_async_to_lds_b128(
      (__attribute__((address_space(1))) i32x4*)g,
      (__attribute__((address_space(3))) i32x4*)l, 0, 0);
#else
  *(uint4*)l = *(const uint4*)g;
#endif
}

static __device__ __forceinline__ void async_wait_all() {
#ifdef HGRN2_ASYNC
  __builtin_amdgcn_s_wait_asynccnt(0);
#endif
}

// A fragment (16x32 bf16, ISA 7.12.2 layout):
//   lanes 0-15 : row = lane,    K in {0..7} U {16..23}
//   lanes 16-31: row = lane-16, K in {8..15} U {24..31}
template <typename T>
static __device__ __forceinline__ bf16x16 frag_a(const T* p, int kstride) {
  const int kb = (laneid() < 16) ? 0 : 8;
  bf16x16 a;
#pragma unroll
  for (int j = 0; j < 8; ++j) a[j] = (__bf16)p[(kb + j) * kstride];
#pragma unroll
  for (int j = 0; j < 8; ++j) a[j + 8] = (__bf16)p[(kb + 16 + j) * kstride];
  return a;
}

// B fragment (32x16 bf16): lane = column; lanes 0-15 K in {0..7} U {16..23}, lanes 16-31 K in {8..15} U {24..31}.
template <typename T>
static __device__ __forceinline__ bf16x16 frag_b(const T* p, int kstride) {
  const int kb = (laneid() < 16) ? 0 : 8;
  bf16x16 b;
#pragma unroll
  for (int j = 0; j < 8; ++j) b[j] = (__bf16)p[(kb + j) * kstride];
#pragma unroll
  for (int j = 0; j < 8; ++j) b[j + 8] = (__bf16)p[(kb + 16 + j) * kstride];
  return b;
}

// ---------------------------------------------------------------------------
// fp32 -> bf16 conversion
// ---------------------------------------------------------------------------
__global__ void hgrn2_f32_to_bf16(const float* __restrict__ s, __bf16* __restrict__ d, size_t n) {
  size_t i = (size_t)blockIdx.x * blockDim.x + threadIdx.x;
  size_t st = (size_t)gridDim.x * blockDim.x;
  for (; i < n; i += st) d[i] = (__bf16)s[i];
}

// ---------------------------------------------------------------------------
// GEMM: out[m,n] = sum_k A[m,k] * W[n,k]  (torch Linear, W row-major [N,K])
// 256 threads = 8 waves; block tile 64x64; wave tile 32(M) x 16(N).
// mode 0: silu(y)*scale -> outb (bf16)                       (q projection)
// mode 1: logsigmoid(y) -> outf (f32) ; 1-sigmoid(y) -> outb (f proj: g, k)
// mode 2: y -> outb (bf16)                                   (v projection)
// mode 3: y -> outf (f32)                                    (final o_proj)
// ---------------------------------------------------------------------------
__global__ __launch_bounds__(256)
void hgrn2_gemm_bf16(const __bf16* __restrict__ A, const __bf16* __restrict__ W,
                     float* __restrict__ outf, __bf16* __restrict__ outb,
                     int N, int K, int mode, float scale)
{
  const int lane = threadIdx.x & 31;
  const int wave = threadIdx.x >> 5;
  const int wm = wave >> 2;                 // 0..1
  const int wn = wave & 3;                  // 0..3
  const int row0 = blockIdx.y * 64 + wm * 32;
  const int col0 = blockIdx.x * 64 + wn * 16;
  const int lrow = lane & 15;
  const int rbase = (lane >= 16) ? 8 : 0;

  floatx8 acc0 = {0.f, 0.f, 0.f, 0.f, 0.f, 0.f, 0.f, 0.f};
  floatx8 acc1 = acc0;

  const __bf16* aRow0 = A + (size_t)(row0 + lrow) * K;
  const __bf16* aRow1 = aRow0 + (size_t)16 * K;
  const __bf16* wRow  = W + (size_t)(col0 + lrow) * K;

  for (int k0 = 0; k0 < K; k0 += 32) {
    bf16x16 bf = frag_b(wRow + k0, 1);
    bf16x16 a0 = frag_a(aRow0 + k0, 1);
    acc0 = WMMA_BF16(a0, bf, acc0);
    bf16x16 a1 = frag_a(aRow1 + k0, 1);
    acc1 = WMMA_BF16(a1, bf, acc1);
  }

  const int colg = col0 + lrow;
#pragma unroll
  for (int m = 0; m < 2; ++m) {
    floatx8 acc = m ? acc1 : acc0;
#pragma unroll
    for (int r = 0; r < 8; ++r) {
      int rowg = row0 + m * 16 + rbase + r;
      size_t idx = (size_t)rowg * N + colg;
      float y = acc[r];
      if (mode == 0) {
        outb[idx] = (__bf16)((y / (1.f + __expf(-y))) * scale);   // silu * DF^-0.5
      } else if (mode == 1) {
        outf[idx] = fminf(y, 0.f) - log1pf(__expf(-fabsf(y)));    // log-sigmoid
        outb[idx] = (__bf16)(1.f / (1.f + __expf(y)));            // 1 - sigmoid(y)
      } else if (mode == 2) {
        outb[idx] = (__bf16)y;
      } else {
        outf[idx] = y;
      }
    }
  }
}

// ---------------------------------------------------------------------------
// Chunked HGRN2 scan. Grid = B*H*4 (state cols d split 4-way, 32 cols each);
// 128 threads = 4 waves. Chunk C=32 (WMMA K). Per chunk:
//   gcum = cumsum(g);  Qg = q*e^gcum;  Kg = k*e^-gcum;  e31 = e^gcum[31]
//   O    = Qg @ S  +  tril(Qg @ Kg^T) @ V
//   S    = e31 (.) (S + Kg^T @ V)
// ---------------------------------------------------------------------------
#define HG_T   4096
#define HG_C   32

__global__ __launch_bounds__(128)
void hgrn2_scan(const __bf16* __restrict__ qb, const float* __restrict__ gs,
                const __bf16* __restrict__ kb, const __bf16* __restrict__ vb,
                float* __restrict__ os)
{
  __shared__ __attribute__((aligned(16))) float  sS[128 * 32];      // S[f][dd] 16 KB
  __shared__ __attribute__((aligned(16))) __bf16 sQ[HG_C * 128];    // Qg[t][f]  8 KB
  __shared__ __attribute__((aligned(16))) __bf16 sK[HG_C * 128];    // Kg[t][f]  8 KB
  __shared__ __attribute__((aligned(16))) __bf16 sV[HG_C * 32];     // V[t][dd]  2 KB
  __shared__ __attribute__((aligned(16))) float  sG[HG_C * 128];    // gcum     16 KB
  __shared__ __attribute__((aligned(16))) float  sAtt[HG_C * HG_C]; // scores    4 KB
  __shared__ __attribute__((aligned(16))) float  sE[128];           // e^gcum31

  const int tid  = threadIdx.x;
  const int lane = tid & 31;
  const int wave = tid >> 5;               // 0..3
  const int wm = wave >> 1;                // time sub-tile 0..1
  const int wn = wave & 1;                 // d sub-tile   0..1
  const int dsl = blockIdx.x & 3;          // d slice
  const int h   = (blockIdx.x >> 2) & 7;
  const int b   = blockIdx.x >> 5;
  const int dbase = dsl * 32;
  const size_t headoff = (size_t)h * 128;
  const size_t base_bt = (size_t)b * HG_T;
  const int lrow = lane & 15;
  const int rbase = (lane >= 16) ? 8 : 0;

  for (int i = tid; i < 128 * 32; i += 128) sS[i] = 0.f;
  __syncthreads();

  for (int ch = 0; ch < HG_T / HG_C; ++ch) {
    const int t0 = ch * HG_C;
    const bool more = (ch + 1) < (HG_T / HG_C);

    // ---- stage chunk into LDS (async DMA when available) ----
    for (int j = tid; j < HG_C * 16; j += 128) {      // q,k: 16x16B per row
      int t = j >> 4, c = (j & 15) * 8;
      size_t ge = (base_bt + t0 + t) * 1024 + headoff + c;
      copy16(qb + ge, sQ + t * 128 + c);
      copy16(kb + ge, sK + t * 128 + c);
      if (more) {
        __builtin_prefetch(qb + ge + (size_t)HG_C * 1024, 0, 1);
        __builtin_prefetch(kb + ge + (size_t)HG_C * 1024, 0, 1);
      }
    }
    for (int j = tid; j < HG_C * 32; j += 128) {      // g (fp32): 32x16B per row
      int t = j >> 5, c = (j & 31) * 4;
      size_t ge = (base_bt + t0 + t) * 1024 + headoff + c;
      copy16(gs + ge, sG + t * 128 + c);
      if (more) __builtin_prefetch(gs + ge + (size_t)HG_C * 1024, 0, 1);
    }
    for (int j = tid; j < HG_C * 4; j += 128) {       // v slice: 4x16B per row
      int t = j >> 2, c = (j & 3) * 8;
      size_t ge = (base_bt + t0 + t) * 1024 + headoff + dbase + c;
      copy16(vb + ge, sV + t * 32 + c);
      if (more) __builtin_prefetch(vb + ge + (size_t)HG_C * 1024, 0, 1);
    }
    async_wait_all();
    __syncthreads();

    // ---- cumulative log-gates; fold decay into q / k ----
    {
      const int f = tid;  // 128 threads == 128 features
      float gc = 0.f, e = 1.f;
      for (int t = 0; t < HG_C; ++t) {
        gc += sG[t * 128 + f];
        sG[t * 128 + f] = gc;
        e = __expf(gc);
        sQ[t * 128 + f] = (__bf16)((float)sQ[t * 128 + f] * e);
        sK[t * 128 + f] = (__bf16)((float)sK[t * 128 + f] / e);
      }
      sE[f] = e;
    }
    __syncthreads();

    // ---- O_inter = Qg @ S  and  Att = Qg @ Kg^T (shared A fragments) ----
    floatx8 oacc = {0.f, 0.f, 0.f, 0.f, 0.f, 0.f, 0.f, 0.f};
    floatx8 at   = oacc;
    for (int kk = 0; kk < 4; ++kk) {
      int fb = kk * 32;
      bf16x16 aQ = frag_a(&sQ[(wm * 16 + lrow) * 128 + fb], 1);
      bf16x16 bS = frag_b(&sS[(size_t)fb * 32 + wn * 16 + lrow], 32);
      oacc = WMMA_BF16(aQ, bS, oacc);
      bf16x16 bK = frag_b(&sK[(wn * 16 + lrow) * 128 + fb], 1);   // Kg^T[f][s]
      at = WMMA_BF16(aQ, bK, at);
    }
#pragma unroll
    for (int r = 0; r < 8; ++r) {
      int tr = wm * 16 + rbase + r;
      int sc = wn * 16 + lrow;
      sAtt[tr * 32 + sc] = (sc <= tr) ? at[r] : 0.f;              // causal mask
    }
    __syncthreads();

    // ---- O += Att @ V ; store O ----
    {
      bf16x16 aA = frag_a(&sAtt[(wm * 16 + lrow) * 32], 1);
      bf16x16 bV = frag_b(&sV[wn * 16 + lrow], 32);
      oacc = WMMA_BF16(aA, bV, oacc);
      int dg = (int)headoff + dbase + wn * 16 + lrow;
#pragma unroll
      for (int r = 0; r < 8; ++r)
        os[(base_bt + t0 + wm * 16 + rbase + r) * 1024 + dg] = oacc[r];
    }

    // ---- state carry: S = e31 (.) (S + Kg^T @ V) ----
#pragma unroll
    for (int i = 0; i < 2; ++i) {
      int f0 = (wave * 2 + i) * 16;
      bf16x16 aK = frag_a(&sK[f0 + lrow], 128);                   // Kg^T[f][t]
#pragma unroll
      for (int nt = 0; nt < 2; ++nt) {
        int dd = nt * 16 + lrow;
        bf16x16 bV2 = frag_b(&sV[dd], 32);
        floatx8 cs;
#pragma unroll
        for (int r = 0; r < 8; ++r) cs[r] = sS[(f0 + rbase + r) * 32 + dd];
        cs = WMMA_BF16(aK, bV2, cs);
#pragma unroll
        for (int r = 0; r < 8; ++r) {
          int f = f0 + rbase + r;
          sS[f * 32 + dd] = sE[f] * cs[r];
        }
      }
    }
    __syncthreads();
  }
}

// ---------------------------------------------------------------------------
// RMSNorm (fp32) * g_weight -> bf16, one block (256 thr) per 1024-row
// ---------------------------------------------------------------------------
__global__ __launch_bounds__(256)
void hgrn2_rmsnorm_bf16(const float* __restrict__ x, const float* __restrict__ w,
                        __bf16* __restrict__ y)
{
  __shared__ float red[8];
  const int row = blockIdx.x;
  const int tid = threadIdx.x;
  const float* xr = x + (size_t)row * 1024;
  float v[4];
  float ss = 0.f;
#pragma unroll
  for (int i = 0; i < 4; ++i) { v[i] = xr[tid + i * 256]; ss += v[i] * v[i]; }
#pragma unroll
  for (int off = 16; off; off >>= 1) ss += __shfl_down(ss, off, 32);
  if ((tid & 31) == 0) red[tid >> 5] = ss;
  __syncthreads();
  float tot = 0.f;
  if (tid < 8) {
    tot = red[tid];
#pragma unroll
    for (int off = 4; off; off >>= 1) tot += __shfl_down(tot, off, 8);
    if (tid == 0) red[0] = tot;
  }
  __syncthreads();
  float inv = rsqrtf(red[0] * (1.f / 1024.f) + 1e-5f);
  __bf16* yr = y + (size_t)row * 1024;
#pragma unroll
  for (int i = 0; i < 4; ++i) {
    int c = tid + i * 256;
    yr[c] = (__bf16)(v[i] * inv * w[c]);
  }
}

// ---------------------------------------------------------------------------
// host launcher
// ---------------------------------------------------------------------------
extern "C" void kernel_launch(void* const* d_in, const int* in_sizes, int n_in,
                              void* d_out, int out_size, void* d_ws, size_t ws_size,
                              hipStream_t stream)
{
  const float* x  = (const float*)d_in[0];
  const float* Wq = (const float*)d_in[1];
  const float* Wf = (const float*)d_in[2];
  const float* Wi = (const float*)d_in[3];
  const float* gw = (const float*)d_in[4];
  const float* Wo = (const float*)d_in[5];
  float* out = (float*)d_out;

  const int D = 1024;
  const size_t M = (size_t)2 * 4096;       // B*T = 8192
  const size_t MD = M * D;
  const size_t DD = (size_t)D * D;

  char* p = (char*)d_ws;
  auto take = [&](size_t bytes) -> void* {
    void* r = (void*)p;
    p += (bytes + 255) & ~(size_t)255;
    return r;
  };
  __bf16* xb  = (__bf16*)take(MD * 2);
  __bf16* wqb = (__bf16*)take(DD * 2);
  __bf16* wfb = (__bf16*)take(DD * 2);
  __bf16* wib = (__bf16*)take(DD * 2);
  __bf16* wob = (__bf16*)take(DD * 2);
  __bf16* qbv = (__bf16*)take(MD * 2);
  __bf16* kbv = (__bf16*)take(MD * 2);
  __bf16* vbv = (__bf16*)take(MD * 2);
  float*  gsv = (float*)take(MD * 4);
  float*  osv = (float*)take(MD * 4);
  __bf16* ob  = (__bf16*)take(MD * 2);

  hgrn2_f32_to_bf16<<<2048, 256, 0, stream>>>(x, xb, MD);
  hgrn2_f32_to_bf16<<<512, 256, 0, stream>>>(Wq, wqb, DD);
  hgrn2_f32_to_bf16<<<512, 256, 0, stream>>>(Wf, wfb, DD);
  hgrn2_f32_to_bf16<<<512, 256, 0, stream>>>(Wi, wib, DD);
  hgrn2_f32_to_bf16<<<512, 256, 0, stream>>>(Wo, wob, DD);

  dim3 gg(D / 64, (unsigned)(M / 64));     // (16, 128)
  const float qscale = 0.08838834764831845f;  // DF^-0.5
  hgrn2_gemm_bf16<<<gg, 256, 0, stream>>>(xb, wqb, nullptr, qbv, D, D, 0, qscale);
  hgrn2_gemm_bf16<<<gg, 256, 0, stream>>>(xb, wfb, gsv,    kbv, D, D, 1, 0.f);
  hgrn2_gemm_bf16<<<gg, 256, 0, stream>>>(xb, wib, nullptr, vbv, D, D, 2, 0.f);

  hgrn2_scan<<<64, 128, 0, stream>>>(qbv, gsv, kbv, vbv, osv);

  hgrn2_rmsnorm_bf16<<<(unsigned)M, 256, 0, stream>>>(osv, gw, ob);

  hgrn2_gemm_bf16<<<gg, 256, 0, stream>>>(ob, wob, out, nullptr, D, D, 3, 0.f);
}
